// CONVKNRM_87385404604468
// MI455X (gfx1250) — matrix operand order repaired and checked by reading, not compiled, not run on hardware
//
#include <hip/hip_runtime.h>
#include <hip/hip_bf16.h>

typedef __bf16 bf16;
typedef __attribute__((ext_vector_type(16))) __bf16 v16bf;
typedef __attribute__((ext_vector_type(8)))  __bf16 bf16x8;
typedef __attribute__((ext_vector_type(8)))  float  v8f;

union FragU { v16bf v; bf16x8 h[2]; };

// A fragment (16x32 bf16, MxK): lane L holds row M=L&15; low half K=kb+{0..7,16..23},
// high half +8.  p must point at (row, kb + (half?8:0)).
__device__ inline v16bf load_a16(const bf16* p) {
    FragU f;
    f.h[0] = *(const bf16x8*)(p);
    f.h[1] = *(const bf16x8*)(p + 16);
    return f.v;
}
// B fragment (32x16 bf16, KxN): lane L holds column N=L&15; low half K=kb+{0..7,16..23},
// high half +8.  p points at (column N's row, kb + (half?8:0)).
__device__ inline v16bf load_b16v(const bf16* p) {
    FragU f;
    f.h[0] = *(const bf16x8*)(p);
    f.h[1] = *(const bf16x8*)(p + 16);
    return f.v;
}

__device__ __constant__ float c_mu[11]  = {1.0f, 0.9f, 0.7f, 0.5f, 0.3f, 0.1f,
                                           -0.1f, -0.3f, -0.5f, -0.7f, -0.9f};
// 1/(2*sigma^2): sigma = 0.001 then 0.1 x10
__device__ __constant__ float c_i2s[11] = {5.0e5f, 50.f, 50.f, 50.f, 50.f, 50.f,
                                           50.f, 50.f, 50.f, 50.f, 50.f};
__device__ __constant__ int c_qt[9] = {0, 0, 0, 1, 2, 1, 1, 2, 2};
__device__ __constant__ int c_dt[9] = {0, 2, 1, 0, 0, 1, 2, 1, 2};

#define EPAD 320   // E=300 padded to multiple of 32
#define CCH  128   // channels

// ---- prep: wv (30000x300 f32) -> wvb (30000x320 bf16, zero padded) ----
__global__ __launch_bounds__(256) void cvt_wv_kernel(const float* __restrict__ wv,
                                                     bf16* __restrict__ wvb, int rows) {
    long i = (long)blockIdx.x * 256 + threadIdx.x;
    long total = (long)rows * EPAD;
    if (i >= total) return;
    int row = (int)(i / EPAD), col = (int)(i % EPAD);
    float v = (col < 300) ? wv[(long)row * 300 + col] : 0.f;
    wvb[i] = (bf16)v;
}

// ---- prep: W (C,k,300) f32 -> Wt (C, 320*k) bf16, channel-major rows ----
__global__ __launch_bounds__(256) void cvt_w_kernel(const float* __restrict__ W,
                                                    bf16* __restrict__ Wt, int kgram) {
    int rowlen = EPAD * kgram;
    long i = (long)blockIdx.x * 256 + threadIdx.x;
    long total = (long)CCH * rowlen;
    if (i >= total) return;
    int c = (int)(i / rowlen);
    int rem = (int)(i % rowlen);
    int j = rem / EPAD, e = rem % EPAD;
    float v = (e < 300) ? W[((long)c * kgram + j) * 300 + e] : 0.f;
    Wt[i] = (bf16)v;
}

// ---- fused gather + n-gram conv GEMM + ReLU + L2norm -> bf16 features ----
// grid.x = nrows/128; block = 256 (8 waves); each wave owns 16 rows x 128 chans
// (8 independent 16x16 f32 accumulators -> 8 WMMAs per 18 b128 loads, and the
//  whole row lives in one wave so the L2-norm is a pure shfl_xor reduction).
template <int KG>
__global__ __launch_bounds__(256)
void conv_wmma_kernel(const bf16* __restrict__ wvb, const int* __restrict__ idx,
                      const bf16* __restrict__ Wt, const float* __restrict__ bias,
                      bf16* __restrict__ outF, int seqlen) {
    const int tid  = threadIdx.x;
    const int wave = tid >> 5;
    const int lane = tid & 31;
    const int m    = lane & 15;
    const int half = lane >> 4;
    const int tileRow = blockIdx.x * 128 + wave * 16;   // 16 | seqlen always
    const int b    = tileRow / seqlen;
    const int l0   = tileRow % seqlen;

    // per-lane token row pointers for each gram word (clamped; tail rows are
    // garbage for k>1 and masked out during pooling)
    int l = l0 + m;
    int lb_ = l + 1, lc_ = l + 2;
    if (lb_ > seqlen - 1) lb_ = seqlen - 1;
    if (lc_ > seqlen - 1) lc_ = seqlen - 1;
    const bf16* rp0 = wvb + (long)idx[(long)b * seqlen + l]   * EPAD + half * 8;
    const bf16* rp1 = (KG > 1) ? wvb + (long)idx[(long)b * seqlen + lb_] * EPAD + half * 8 : rp0;
    const bf16* rp2 = (KG > 2) ? wvb + (long)idx[(long)b * seqlen + lc_] * EPAD + half * 8 : rp0;

    const int rowlen = EPAD * KG;
    const bf16* wbase = Wt + (long)m * rowlen + half * 8;  // lane's channel column m

    v8f acc[8] = {};
#pragma unroll
    for (int j = 0; j < KG; ++j) {
        const bf16* rpj = (j == 0) ? rp0 : ((j == 1) ? rp1 : rp2);
#pragma unroll
        for (int kc = 0; kc < EPAD / 32; ++kc) {
            v16bf a = load_a16(rpj + kc * 32);
#pragma unroll
            for (int jj = 0; jj < 8; ++jj) {
                v16bf bm = load_b16v(wbase + (long)jj * 16 * rowlen + j * EPAD + kc * 32);
                acc[jj] = __builtin_amdgcn_wmma_f32_16x16x32_bf16(
                    false, a, false, bm, (short)0, acc[jj], false, false);
            }
        }
    }

    // epilogue: bias + ReLU + 1e-9, in-wave row L2-norm over all 128 channels.
    float bb[8];
#pragma unroll
    for (int jj = 0; jj < 8; ++jj) bb[jj] = bias[16 * jj + m];

#pragma unroll
    for (int r = 0; r < 8; ++r) {
        float t = 0.f;
#pragma unroll
        for (int jj = 0; jj < 8; ++jj) {
            float x = acc[jj][r] + bb[jj];
            x = x > 0.f ? x : 0.f;
            x += 1e-9f;
            t += x * x;
        }
        t += __shfl_xor(t, 1, 32);
        t += __shfl_xor(t, 2, 32);
        t += __shfl_xor(t, 4, 32);
        t += __shfl_xor(t, 8, 32);      // full 128-col sum for row r + 8*half
        float sc = 1.0f / fmaxf(sqrtf(t), 1e-10f);
        long row = tileRow + r + 8 * half;
#pragma unroll
        for (int jj = 0; jj < 8; ++jj) {
            float x = acc[jj][r] + bb[jj];
            x = x > 0.f ? x : 0.f;
            x += 1e-9f;
            outF[row * CCH + 16 * jj + m] = (bf16)(x * sc);
        }
    }
}

// ---- similarity WMMA + Gaussian kernel pooling (partials per chunk) ----
// grid.x = B*9*8 ; block = 256.  Each iteration: 8 waves build a 16x128 sim
// strip in LDS via 16x16x32 bf16 WMMA, then 176 threads (q,bin) pool it.
__global__ __launch_bounds__(256)
void simpool_kernel(const bf16* __restrict__ fq, const bf16* __restrict__ fd,
                    float* __restrict__ partial) {
    __shared__ float simLDS[16][132];          // padded vs bank conflicts
    const int gid   = blockIdx.x;
    const int chunk = gid & 7;
    const int bp    = gid >> 3;
    const int pair  = bp % 9;
    const int b     = bp / 9;
    const int qt = c_qt[pair], dt = c_dt[pair];
    const int Qp = 16 - qt;                    // valid query positions
    const int Dp = 4096 - dt;                  // valid doc positions
    const bf16* fqp = fq + (long)qt * (512L * CCH) + (long)b * 16 * CCH;
    const bf16* fdp = fd + (long)dt * (131072L * CCH) + (long)b * 4096 * CCH;

    const int tid = threadIdx.x;
    const int wave = tid >> 5, lane = tid & 31;
    const int m = lane & 15, half = lane >> 4;
    const int tq = tid / 11, tb = tid % 11;    // pooling assignment (tid<176)
    const float mu_t  = c_mu[tb < 11 ? tb : 0];
    const float i2s_t = c_i2s[tb < 11 ? tb : 0];

    const bf16* arow = fqp + (long)m * CCH + half * 8;   // per-lane query row
    float poolacc = 0.f;

    for (int it = 0; it < 4; ++it) {
        const int dbase = chunk * 512 + it * 128;
        const int dcol  = dbase + wave * 16 + m;
        const bf16* brow = fdp + (long)dcol * CCH + half * 8;
        v8f acc = {};
#pragma unroll
        for (int kb = 0; kb < CCH; kb += 32) {
            v16bf a  = load_a16(arow + kb);
            v16bf bm = load_b16v(brow + kb);
            acc = __builtin_amdgcn_wmma_f32_16x16x32_bf16(
                false, a, false, bm, (short)0, acc, false, false);
        }
#pragma unroll
        for (int r = 0; r < 8; ++r)
            simLDS[r + 8 * half][wave * 16 + m] = acc[r];
        __syncthreads();
        if (tid < 176 && tq < Qp) {
            int dmax = Dp - dbase;
            if (dmax > 128) dmax = 128;
            for (int dd = 0; dd < dmax; ++dd) {
                float s  = simLDS[tq][dd];
                float df = s - mu_t;
                poolacc += __expf(-df * df * i2s_t);
            }
        }
        __syncthreads();
    }
    if (tid < 176)
        partial[(long)gid * 176 + tid] = poolacc;
}

// ---- finalize: sum chunks, log-clip, reduce over q -> out (32,99) f32 ----
__global__ __launch_bounds__(192)
void finalize_kernel(const float* __restrict__ partial, float* __restrict__ out) {
    __shared__ float red[176];
    const int bp = blockIdx.x;
    const int pair = bp % 9, b = bp / 9;
    const int Qp = 16 - c_qt[pair];
    const int t = threadIdx.x;
    if (t < 176) {
        float s = 0.f;
        for (int c = 0; c < 8; ++c)
            s += partial[((long)bp * 8 + c) * 176 + t];
        const int q = t / 11;
        red[t] = (q < Qp) ? 0.01f * logf(fmaxf(s, 1e-10f)) : 0.f;
    }
    __syncthreads();
    if (t < 11) {
        float o = 0.f;
        for (int q = 0; q < 16; ++q) o += red[q * 11 + t];
        out[(long)b * 99 + pair * 11 + t] = o;
    }
}

extern "C" void kernel_launch(void* const* d_in, const int* in_sizes, int n_in,
                              void* d_out, int out_size, void* d_ws, size_t ws_size,
                              hipStream_t stream) {
    const int*   q_idx = (const int*)d_in[0];    // (32,16)
    const int*   d_idx = (const int*)d_in[1];    // (32,4096)
    const float* wv    = (const float*)d_in[3];  // (30000,300)
    const float* W_u   = (const float*)d_in[4];
    const float* b_u   = (const float*)d_in[5];
    const float* W_b   = (const float*)d_in[6];
    const float* b_b   = (const float*)d_in[7];
    const float* W_t   = (const float*)d_in[8];
    const float* b_t   = (const float*)d_in[9];
    float* out = (float*)d_out;

    const int V = 30000;
    char* ws = (char*)d_ws;
    size_t off = 0;
    bf16* wvb = (bf16*)(ws + off);   off += (size_t)V * EPAD * 2;        // 19.2 MB
    bf16* Wtu = (bf16*)(ws + off);   off += (size_t)CCH * EPAD * 2;      // 80 KB
    bf16* Wtb = (bf16*)(ws + off);   off += (size_t)CCH * EPAD * 2 * 2;  // 160 KB
    bf16* Wtt = (bf16*)(ws + off);   off += (size_t)CCH * EPAD * 2 * 3;  // 240 KB
    bf16* fq  = (bf16*)(ws + off);   off += 3L * 512 * CCH * 2;          // 384 KB
    bf16* fd  = (bf16*)(ws + off);   off += 3L * 131072 * CCH * 2;       // 96 MB
    float* part = (float*)(ws + off);                                    // 1.55 MB
    (void)ws_size; (void)in_sizes; (void)n_in; (void)out_size;

    // 1) precision re-encode (L2-resident bf16 operands for WMMA)
    cvt_wv_kernel<<<(V * EPAD + 255) / 256, 256, 0, stream>>>(wv, wvb, V);
    cvt_w_kernel<<<(CCH * EPAD * 1 + 255) / 256, 256, 0, stream>>>(W_u, Wtu, 1);
    cvt_w_kernel<<<(CCH * EPAD * 2 + 255) / 256, 256, 0, stream>>>(W_b, Wtb, 2);
    cvt_w_kernel<<<(CCH * EPAD * 3 + 255) / 256, 256, 0, stream>>>(W_t, Wtt, 3);

    // 2) fused gather+conv+relu+l2norm (queries: 512 rows; docs: 131072 rows)
    conv_wmma_kernel<1><<<4, 256, 0, stream>>>(wvb, q_idx, Wtu, b_u, fq + 0L * 512 * CCH, 16);
    conv_wmma_kernel<2><<<4, 256, 0, stream>>>(wvb, q_idx, Wtb, b_b, fq + 1L * 512 * CCH, 16);
    conv_wmma_kernel<3><<<4, 256, 0, stream>>>(wvb, q_idx, Wtt, b_t, fq + 2L * 512 * CCH, 16);
    conv_wmma_kernel<1><<<1024, 256, 0, stream>>>(wvb, d_idx, Wtu, b_u, fd + 0L * 131072 * CCH, 4096);
    conv_wmma_kernel<2><<<1024, 256, 0, stream>>>(wvb, d_idx, Wtb, b_b, fd + 1L * 131072 * CCH, 4096);
    conv_wmma_kernel<3><<<1024, 256, 0, stream>>>(wvb, d_idx, Wtt, b_t, fd + 2L * 131072 * CCH, 4096);

    // 3) similarity WMMA + Gaussian pooling partials, then finalize
    simpool_kernel<<<32 * 9 * 8, 256, 0, stream>>>(fq, fd, part);
    finalize_kernel<<<32 * 9, 192, 0, stream>>>(part, out);
}
